// EM_BiLSTM_CRF_67920612819687
// MI455X (gfx1250) — compile-verified, matrix operand order checked by reading, not run on hardware
//
#include <hip/hip_runtime.h>
#include <cstdint>

typedef __attribute__((ext_vector_type(16))) _Float16 v16h;
typedef __attribute__((ext_vector_type(8)))  _Float16 v8h;
typedef __attribute__((ext_vector_type(8)))  float    v8f;
typedef unsigned int u32x4 __attribute__((ext_vector_type(4)));
typedef int          i32x4 __attribute__((ext_vector_type(4)));
typedef int          i32x8 __attribute__((ext_vector_type(8)));

#define Tt 512
#define Bb 64
#define Ee 256
#define Hh 256
#define G4 1024   // 4*H
#define HD 512
#define Ff 128
#define Vv 50000

// ---------------------------------------------------------------------------
// WMMA tile helpers (wave32, v_wmma_f32_16x16x32_f16 layouts)
// A 16x32 f16: lane 0-15 -> M=lane, K in {0..7,16..23}; lane 16-31 -> M=lane-16,
// K in {8..15,24..31}. B 32x16: symmetric with N per lane. C/D f32: lane ->
// N=lane&15, vgpr j -> M = j + (lane>=16 ? 8 : 0).
// All f16 sources are 16B aligned -> two b128 loads per tile per lane.
// ---------------------------------------------------------------------------
__device__ inline v16h tile_f16(const _Float16* src, int ld, int lane) {
  const int m = lane & 15, half = (lane >> 4) << 3;
  const v8h a0 = *(const v8h*)(src + m * ld + half);
  const v8h a1 = *(const v8h*)(src + m * ld + 16 + half);
  return __builtin_shufflevector(a0, a1, 0, 1, 2, 3, 4, 5, 6, 7,
                                 8, 9, 10, 11, 12, 13, 14, 15);
}

__device__ inline v8f acc_from_f16(const _Float16* src, int ld, int lane) {
  const int n = lane & 15, mb = (lane >> 4) << 3;
  v8f c;
#pragma unroll
  for (int j = 0; j < 8; ++j) c[j] = (float)src[(mb + j) * ld + n];
  return c;
}

__device__ inline void acc_store_f16(_Float16* dst, int ld, int lane, v8f c, float bias) {
  const int n = lane & 15, mb = (lane >> 4) << 3;
#pragma unroll
  for (int j = 0; j < 8; ++j) dst[(mb + j) * ld + n] = (_Float16)(c[j] + bias);
}

__device__ inline float sigmoidf(float x) { return 1.0f / (1.0f + __expf(-x)); }

// ---------------------------------------------------------------------------
// Tensor Data Mover: stage one contiguous 16x1024-f16 tile (32 KB) into LDS.
// D# per CDNA5 ISA ch.10.8: group0 = {count/type/lds_addr/global_addr},
// group1 = {data_size, tensor dims, tile dims, strides}. 1-D view:
// tile_dim0 = tensor_dim0 = stride0 = 16384 elements of 2 bytes.
// ---------------------------------------------------------------------------
__device__ inline void tdm_load_pre_tile(unsigned lds_addr, const _Float16* gptr) {
  const unsigned long long ga = (unsigned long long)(uintptr_t)gptr;
  const unsigned td0 = 16u * 1024u;   // elements (2B each)
  u32x4 g0;
  g0[0] = 1u;                                            // count=1, user mode
  g0[1] = lds_addr;                                      // lds_addr[31:0]
  g0[2] = (unsigned)(ga & 0xffffffffu);                  // global_addr[31:0]
  g0[3] = (unsigned)((ga >> 32) & 0x01ffffffu) | (2u << 30);  // addr[56:32] | type=2
  i32x8 g1;
  g1[0] = (1 << 16);                                     // data_size=1 (2 bytes)
  g1[1] = (int)((td0 & 0xffffu) << 16);                  // tensor_dim0[15:0]
  g1[2] = (int)((td0 >> 16) | (1u << 16));               // tensor_dim0[31:16] | tensor_dim1=1
  g1[3] = (int)(td0 << 16);                              // tile_dim0 = 16384
  g1[4] = 1;                                             // tile_dim1=1, tile_dim2=0
  g1[5] = (int)td0;                                      // tensor_dim0_stride[31:0]
  g1[6] = 0;
  g1[7] = 0;
  const i32x4 z4 = {0, 0, 0, 0};
#if __clang_major__ >= 23
  const i32x8 z8 = {0, 0, 0, 0, 0, 0, 0, 0};
  __builtin_amdgcn_tensor_load_to_lds(g0, g1, z4, z4, z8, 0);
#else
  __builtin_amdgcn_tensor_load_to_lds(g0, g1, z4, z4, 0);
#endif
}

// ---------------------------------------------------------------------------
// Kernel 0: generic f32 -> f16 conversion (one-time weight/embedding prep)
// ---------------------------------------------------------------------------
__global__ void cvt_f16_kernel(const float* __restrict__ src,
                               _Float16* __restrict__ dst, long n) {
  long i = (long)blockIdx.x * blockDim.x + threadIdx.x;
  if (i < n) dst[i] = (_Float16)src[i];
}

// ---------------------------------------------------------------------------
// Kernel 1: fused embedding gather + input projection (both directions)
// pre[dir][t*B + b][g] = sum_e embed[sent[b][t]][e] * Wih[g][e] + bih[g]+bhh[g]
// One wave per 16x16 output tile, K = E = 256 (8 WMMAs).
// ---------------------------------------------------------------------------
__global__ __launch_bounds__(128) void embed_project_kernel(
    const int* __restrict__ sent, const _Float16* __restrict__ embed16,
    const _Float16* __restrict__ wih16,  // [2][1024][256]
    const float* __restrict__ bih_f, const float* __restrict__ bhh_f,
    const float* __restrict__ bih_b, const float* __restrict__ bhh_b,
    _Float16* __restrict__ pre) {
  const int lane = threadIdx.x & 31, wave = threadIdx.x >> 5;
  const int mt = blockIdx.x;                     // 0..2047 (T*B/16)
  const int ntile = blockIdx.y * 4 + wave;       // 0..127  (2*G4/16)
  const int dir = ntile >> 6;
  const int g0 = (ntile & 63) << 4;
  const _Float16* W = wih16 + (long)dir * G4 * Ee;
  const float* bih = dir ? bih_b : bih_f;
  const float* bhh = dir ? bhh_b : bhh_f;

  const int r0 = mt << 4;
  const int m = lane & 15, half = (lane >> 4) << 3;
  const int r = r0 + m;
  const int t = r >> 6, b = r & 63;              // r = t*B + b
  const _Float16* arow = embed16 + (long)sent[b * Tt + t] * Ee;

  v8f acc = {};
  for (int k0 = 0; k0 < Ee; k0 += 32) {
    const v8h a0 = *(const v8h*)(arow + k0 + half);
    const v8h a1 = *(const v8h*)(arow + k0 + 16 + half);
    v16h a = __builtin_shufflevector(a0, a1, 0, 1, 2, 3, 4, 5, 6, 7,
                                     8, 9, 10, 11, 12, 13, 14, 15);
    v16h bm = tile_f16(W + (long)g0 * Ee + k0, Ee, lane);
    acc = __builtin_amdgcn_wmma_f32_16x16x32_f16(false, a, false, bm,
                                                 (short)0, acc, false, false);
  }
  const int gcol = g0 + (lane & 15);
  const float bias = bih[gcol] + bhh[gcol];
  acc_store_f16(pre + ((long)dir * Tt * Bb + r0) * G4 + g0, G4, lane, acc, bias);
}

// ---------------------------------------------------------------------------
// Kernel 2: LSTM recurrence. 8 workgroups: dir = blk>>2, 16 batch rows each.
// 16 waves; wave w owns hidden columns [w*16, w*16+16), computes all 4 gate
// tiles. pre[t] staged into LDS by the Tensor Data Mover, one step ahead.
// h double-buffered in LDS (f16); cell state c lives in registers (lane/vgpr
// <-> (m,n) mapping of the D tile is time-invariant).
// ---------------------------------------------------------------------------
__global__ __launch_bounds__(512) void lstm_rec_kernel(
    const _Float16* __restrict__ pre, const _Float16* __restrict__ whh16,
    _Float16* __restrict__ hs) {
  __shared__ _Float16 prelds[16][G4];   // 32 KB, TDM destination
  __shared__ _Float16 hbuf[2][16][Hh];  // 16 KB

  const int dir = blockIdx.x >> 2;
  const int mt  = blockIdx.x & 3;                // batch tile (16 rows)
  const int lane = threadIdx.x & 31, wave = threadIdx.x >> 5;
  const int n0 = wave << 4;
  const int n  = lane & 15, mb = (lane >> 4) << 3;

  for (int i = threadIdx.x; i < 16 * Hh; i += 512) {
    int mm = i >> 8, kk = i & 255;
    hbuf[0][mm][kk] = (_Float16)0.0f;
    hbuf[1][mm][kk] = (_Float16)0.0f;
  }
  float creg[8];
#pragma unroll
  for (int j = 0; j < 8; ++j) creg[j] = 0.0f;

  const _Float16* predir = pre + (long)dir * Tt * Bb * G4;
  const _Float16* whhdir = whh16 + (long)dir * G4 * Hh;
  const unsigned prelds_addr = (unsigned)(uintptr_t)&prelds[0][0];

  // stage pre[t(0)]
  if (threadIdx.x == 0) {
    const int t0 = dir ? (Tt - 1) : 0;
    tdm_load_pre_tile(prelds_addr, predir + ((long)t0 * Bb + mt * 16) * G4);
  }
  __syncthreads();

  for (int s = 0; s < Tt; ++s) {
    const int t = dir ? (Tt - 1 - s) : s;
    const int cur = s & 1, nxt = cur ^ 1;

    if (threadIdx.x == 0) __builtin_amdgcn_s_wait_tensorcnt(0);
    __syncthreads();  // prelds holds pre[t]; hbuf[cur] holds h[t-1]

    v8f acc[4];
#pragma unroll
    for (int g = 0; g < 4; ++g)
      acc[g] = acc_from_f16(&prelds[0][g * Hh + n0], G4, lane);

    __syncthreads();  // all waves done reading prelds -> safe to overwrite
    if (s + 1 < Tt && threadIdx.x == 0) {
      const int tn = dir ? (Tt - 2 - s) : (s + 1);
      tdm_load_pre_tile(prelds_addr, predir + ((long)tn * Bb + mt * 16) * G4);
    }

    for (int k0 = 0; k0 < Hh; k0 += 32) {
      v16h a = tile_f16(&hbuf[cur][0][k0], Hh, lane);
#pragma unroll
      for (int g = 0; g < 4; ++g) {
        v16h bm = tile_f16(whhdir + (long)(g * Hh + n0) * Hh + k0, Hh, lane);
        acc[g] = __builtin_amdgcn_wmma_f32_16x16x32_f16(false, a, false, bm,
                                                        (short)0, acc[g], false, false);
      }
    }

    const int hidx = n0 + n;
#pragma unroll
    for (int j = 0; j < 8; ++j) {
      const int m = mb + j;
      const float iv = sigmoidf(acc[0][j]);
      const float fv = sigmoidf(acc[1][j]);
      const float gv = tanhf(acc[2][j]);
      const float ov = sigmoidf(acc[3][j]);
      const float c = fv * creg[j] + iv * gv;
      creg[j] = c;
      const float h = ov * tanhf(c);
      hbuf[nxt][m][hidx] = (_Float16)h;
      hs[((long)t * Bb + mt * 16 + m) * HD + dir * Hh + hidx] = (_Float16)h;
    }
    // loop-top barrier orders hbuf[nxt] writes before next step's reads
  }
}

// ---------------------------------------------------------------------------
// Kernel 3: feats = concat(hf,hb) @ W_lin^T + b_lin   (M=32768, N=128, K=512)
// ---------------------------------------------------------------------------
__global__ __launch_bounds__(128) void linear_kernel(
    const _Float16* __restrict__ hs, const _Float16* __restrict__ wlin16,
    const float* __restrict__ blin, _Float16* __restrict__ feats) {
  const int lane = threadIdx.x & 31, wave = threadIdx.x >> 5;
  const int r0 = blockIdx.x << 4;
  const int f0 = (blockIdx.y * 4 + wave) << 4;
  v8f acc = {};
  for (int k0 = 0; k0 < HD; k0 += 32) {
    v16h a  = tile_f16(hs + (long)r0 * HD + k0, HD, lane);
    v16h bm = tile_f16(wlin16 + (long)f0 * HD + k0, HD, lane);
    acc = __builtin_amdgcn_wmma_f32_16x16x32_f16(false, a, false, bm,
                                                 (short)0, acc, false, false);
  }
  acc_store_f16(feats + (long)r0 * Ff + f0, Ff, lane, acc, blin[f0 + (lane & 15)]);
}

// ---------------------------------------------------------------------------
// Kernel 4: em2 = feats @ W2^T + b2   (M=32768, N=32, K=128), WMMA.
// ---------------------------------------------------------------------------
__global__ __launch_bounds__(64) void emit2_kernel(
    const _Float16* __restrict__ feats, const _Float16* __restrict__ w2_16,
    const float* __restrict__ b2, float* __restrict__ em2) {
  const int lane = threadIdx.x & 31, wave = threadIdx.x >> 5;
  const int r0 = blockIdx.x << 4;
  const int n0 = wave << 4;
  v8f acc = {};
  for (int k0 = 0; k0 < Ff; k0 += 32) {
    v16h a  = tile_f16(feats + (long)r0 * Ff + k0, Ff, lane);
    v16h bm = tile_f16(w2_16 + (long)n0 * Ff + k0, Ff, lane);
    acc = __builtin_amdgcn_wmma_f32_16x16x32_f16(false, a, false, bm,
                                                 (short)0, acc, false, false);
  }
  const int nn = lane & 15, mbase = (lane >> 4) << 3;
  const float bias = b2[n0 + nn];
#pragma unroll
  for (int j = 0; j < 8; ++j)
    em2[(long)(r0 + mbase + j) * 32 + n0 + nn] = acc[j] + bias;
}

// ---------------------------------------------------------------------------
// Kernel 5: em1 = feats @ W1^T + b1  (N=4 — too skinny for WMMA, plain VALU)
// ---------------------------------------------------------------------------
__global__ void emit1_kernel(const _Float16* __restrict__ feats,
                             const float* __restrict__ w1,
                             const float* __restrict__ b1,
                             float* __restrict__ em1) {
  const int r = blockIdx.x * blockDim.x + threadIdx.x;
  if (r >= Tt * Bb) return;
  const _Float16* fr = feats + (long)r * Ff;
#pragma unroll
  for (int k = 0; k < 4; ++k) {
    float s = b1[k];
    const float* wr = w1 + k * Ff;
    for (int j = 0; j < Ff; ++j) s += (float)fr[j] * wr[j];
    em1[(long)r * 4 + k] = s;
  }
}

// ---------------------------------------------------------------------------
// Kernel 6: Viterbi decode, one wave per batch row. em layout [T][B][K].
// ---------------------------------------------------------------------------
__global__ __launch_bounds__(32) void viterbi_kernel(
    const float* __restrict__ em, const float* __restrict__ start,
    const float* __restrict__ trans, const float* __restrict__ endv, int K,
    unsigned char* __restrict__ hist, float* __restrict__ out) {
  const int b = blockIdx.x;
  const int lane = threadIdx.x;
  __shared__ float tr[32 * 32];
  __shared__ float sc[32];
  for (int i = lane; i < K * K; i += 32) tr[i] = trans[i];
  sc[lane] = (lane < K) ? start[lane] + em[(long)b * K + lane] : -3e38f;
  __syncthreads();
  for (int t = 1; t < Tt; ++t) {
    float best = -3e38f;
    int arg = 0;
    if (lane < K) {
      for (int p = 0; p < K; ++p) {
        float v = sc[p] + tr[p * K + lane];
        if (v > best) { best = v; arg = p; }
      }
      best += em[((long)t * Bb + b) * K + lane];
    }
    __syncthreads();
    sc[lane] = (lane < K) ? best : -3e38f;
    hist[((long)t * Bb + b) * 32 + lane] = (unsigned char)arg;
    __syncthreads();
  }
  if (lane < K) sc[lane] += endv[lane];
  __syncthreads();
  if (lane == 0) {
    int tag = 0;
    float bv = sc[0];
    for (int k = 1; k < K; ++k) if (sc[k] > bv) { bv = sc[k]; tag = k; }
    out[(long)b * Tt + Tt - 1] = (float)tag;
    for (int t = Tt - 1; t >= 1; --t) {
      tag = hist[((long)t * Bb + b) * 32 + tag];
      out[(long)b * Tt + t - 1] = (float)tag;
    }
  }
}

// ---------------------------------------------------------------------------
extern "C" void kernel_launch(void* const* d_in, const int* in_sizes, int n_in,
                              void* d_out, int out_size, void* d_ws, size_t ws_size,
                              hipStream_t stream) {
  const int*   sentences = (const int*)d_in[0];
  const float* embed  = (const float*)d_in[3];
  const float* wih_f  = (const float*)d_in[4];
  const float* whh_f  = (const float*)d_in[5];
  const float* bih_f  = (const float*)d_in[6];
  const float* bhh_f  = (const float*)d_in[7];
  const float* wih_b  = (const float*)d_in[8];
  const float* whh_b  = (const float*)d_in[9];
  const float* bih_b  = (const float*)d_in[10];
  const float* bhh_b  = (const float*)d_in[11];
  const float* wlin   = (const float*)d_in[12];
  const float* blin   = (const float*)d_in[13];
  const float* w1     = (const float*)d_in[14];
  const float* b1     = (const float*)d_in[15];
  const float* w2     = (const float*)d_in[16];
  const float* b2     = (const float*)d_in[17];
  const float* start1 = (const float*)d_in[18];
  const float* end1   = (const float*)d_in[19];
  const float* trans1 = (const float*)d_in[20];
  const float* start2 = (const float*)d_in[21];
  const float* end2   = (const float*)d_in[22];
  const float* trans2 = (const float*)d_in[23];

  // Workspace layout (all segment sizes are multiples of 256B)
  char* ws = (char*)d_ws;
  size_t off = 0;
  _Float16* embed16 = (_Float16*)(ws + off); off += (size_t)Vv * Ee * 2;            // 25.6 MB
  _Float16* wih16   = (_Float16*)(ws + off); off += (size_t)2 * G4 * Ee * 2;        // 1 MB
  _Float16* whh16   = (_Float16*)(ws + off); off += (size_t)2 * G4 * Hh * 2;        // 1 MB
  _Float16* wlin16  = (_Float16*)(ws + off); off += (size_t)Ff * HD * 2;            // 128 KB
  _Float16* w2_16   = (_Float16*)(ws + off); off += (size_t)32 * Ff * 2;            // 8 KB
  _Float16* pre     = (_Float16*)(ws + off); off += (size_t)2 * Tt * Bb * G4 * 2;   // 134 MB
  _Float16* hs      = (_Float16*)(ws + off); off += (size_t)Tt * Bb * HD * 2;       // 33.5 MB
  _Float16* feats   = (_Float16*)(ws + off); off += (size_t)Tt * Bb * Ff * 2;       // 8.4 MB
  float* em2        = (float*)(ws + off);    off += (size_t)Tt * Bb * 32 * 4;       // 4.2 MB
  float* em1        = (float*)(ws + off);    off += (size_t)Tt * Bb * 4 * 4;        // 0.5 MB
  unsigned char* hist1 = (unsigned char*)(ws + off); off += (size_t)Tt * Bb * 32;   // 1 MB
  unsigned char* hist2 = (unsigned char*)(ws + off); off += (size_t)Tt * Bb * 32;   // 1 MB

  float* out = (float*)d_out;  // [dec_div (B*T) | dec_ch (B*T)]

  // 0) one-time f32 -> f16 conversion of embeddings + GEMM weights
  auto cvt = [&](const float* s, _Float16* d, long n) {
    cvt_f16_kernel<<<(unsigned)((n + 255) / 256), 256, 0, stream>>>(s, d, n);
  };
  cvt(embed, embed16, (long)Vv * Ee);
  cvt(wih_f, wih16, (long)G4 * Ee);
  cvt(wih_b, wih16 + (size_t)G4 * Ee, (long)G4 * Ee);
  cvt(whh_f, whh16, (long)G4 * Hh);
  cvt(whh_b, whh16 + (size_t)G4 * Hh, (long)G4 * Hh);
  cvt(wlin, wlin16, (long)Ff * HD);
  cvt(w2, w2_16, (long)32 * Ff);

  // 1) embedding gather + input projection (both directions)
  embed_project_kernel<<<dim3(Tt * Bb / 16, 32), 128, 0, stream>>>(
      sentences, embed16, wih16, bih_f, bhh_f, bih_b, bhh_b, pre);

  // 2) recurrence: 8 WGs (2 dirs x 4 batch tiles), 16 waves each, TDM staging
  lstm_rec_kernel<<<8, 512, 0, stream>>>(pre, whh16, hs);

  // 3) linear: feats = hs @ W_lin^T + b_lin
  linear_kernel<<<dim3(Tt * Bb / 16, 2), 128, 0, stream>>>(hs, wlin16, blin, feats);

  // 4) emissions
  emit2_kernel<<<Tt * Bb / 16, 64, 0, stream>>>(feats, w2_16, b2, em2);
  emit1_kernel<<<(Tt * Bb + 127) / 128, 128, 0, stream>>>(feats, w1, b1, em1);

  // 5) Viterbi decodes: dec_div (K1=4) first, then dec_ch (K2=32)
  viterbi_kernel<<<Bb, 32, 0, stream>>>(em1, start1, trans1, end1, 4, hist1, out);
  viterbi_kernel<<<Bb, 32, 0, stream>>>(em2, start2, trans2, end2, 32, hist2,
                                        out + (size_t)Bb * Tt);
}
